// MLPAsGNN_9663676416892
// MI455X (gfx1250) — matrix operand order repaired and checked by reading, not compiled, not run on hardware
//
#include <hip/hip_runtime.h>

// ---------------------------------------------------------------------------
// Fused 2-layer MLP (token-wise) for MI455X / gfx1250, bf16 WMMA.
//
// Shapes: x[32768,1024] f32, W1[4096,1024], b1[4096], W2[1024,4096], b2[1024].
// out[32768,1024] f32.
//
// Memory-bound (~285 MB min HBM vs 0.55 TFLOP). Strategy:
//  * fuse both GEMMs: hidden activations never touch HBM
//  * block the hidden dim (8 blocks of 512): h_blk lives in LDS, the output
//    tile accumulates in registers across blocks -> MT=64 tokens/WG, which
//    halves L2 weight traffic (512 WGs x 16 MB) and gives 4 WMMAs per 32 B
//    weight fragment load.
//  * LDS fragments interleaved [kblk][sub] so all subtile A-frags of one
//    k-step are reachable via DS immediate offsets from one address register.
// ---------------------------------------------------------------------------

typedef __attribute__((ext_vector_type(16))) __bf16 v16bf;
typedef __attribute__((ext_vector_type(4)))  __bf16 v4bf;
typedef __attribute__((ext_vector_type(8)))  float  v8f;
typedef __attribute__((ext_vector_type(4)))  float  v4f;

#define IN_CH   1024
#define HID_CH  4096
#define OUT_CH  1024
#define MT      64            // tokens per workgroup
#define SUB     4             // MT/16 WMMA row subtiles
#define HB      512           // hidden block width
#define NHB     (HID_CH / HB) // 8
#define NWAVE   8             // 256 threads, wave32

// LDS (dynamic, 192 KB):
//   xs:    x A-fragments,      [32 kblk][4 sub][32 lanes][32 B] = 128 KB
//   h_blk: hidden A-fragments, [16 kblk][4 sub][32 lanes][32 B] =  64 KB
#define FRAG_B       (32 * 32)                       // 1024 B per 16x32 frag
#define XS_BYTES     (32 * SUB * FRAG_B)             // 131072
#define HBK_OFF      XS_BYTES
#define LDS_BYTES    (HBK_OFF + (HB / 32) * SUB * FRAG_B)  // 196608

__device__ __forceinline__ __bf16 tobf(float f) {
  // round-to-nearest-even f32 -> bf16
  unsigned u = __builtin_bit_cast(unsigned, f);
  u += 0x7fffu + ((u >> 16) & 1u);
  unsigned short s = (unsigned short)(u >> 16);
  return __builtin_bit_cast(__bf16, s);
}

__device__ __forceinline__ v8f wmma_bf16(v16bf a, v16bf b, v8f c) {
  return __builtin_amdgcn_wmma_f32_16x16x32_bf16(
      /*neg_a=*/false, a, /*neg_b=*/false, b,
      /*c_mod=*/(short)0, c, /*reuse_a=*/false, /*reuse_b=*/false);
}

// ---------------------------------------------------------------------------
// One-shot f32 -> bf16 conversion of both weight matrices (same element count).
// ---------------------------------------------------------------------------
__global__ __launch_bounds__(256) void mlp_cvt_weights(
    const float* __restrict__ w1, const float* __restrict__ w2,
    __bf16* __restrict__ o1, __bf16* __restrict__ o2) {
  size_t i = ((size_t)blockIdx.x * 256 + threadIdx.x) * 4;
  v4f a = *(const v4f*)(w1 + i);
  v4f b = *(const v4f*)(w2 + i);
  v4bf oa = { tobf(a.x), tobf(a.y), tobf(a.z), tobf(a.w) };
  v4bf ob = { tobf(b.x), tobf(b.y), tobf(b.z), tobf(b.w) };
  *(v4bf*)(o1 + i) = oa;
  *(v4bf*)(o2 + i) = ob;
}

// ---------------------------------------------------------------------------
// Fused MLP kernel. Grid: M/MT blocks of 256 threads (8 waves).
// ---------------------------------------------------------------------------
__global__ __launch_bounds__(256, 1) void mlp_fused(
    const float* __restrict__ x,
    const __bf16* __restrict__ w1b, const float* __restrict__ b1,
    const __bf16* __restrict__ w2b, const float* __restrict__ b2,
    float* __restrict__ out) {
  extern __shared__ char smem[];

  const int tid   = threadIdx.x;
  const int lane  = tid & 31;
  const int wv    = tid >> 5;
  const int bcol  = lane & 15;   // WMMA B/C column within an N-tile
  const int bg    = lane >> 4;   // K-group (B) / row group (C)
  const int mbase = blockIdx.x * MT;

  const v8f vzero = {};
  // Persistent output accumulators: wave owns out cols ot = wv + 8*j, all SUB
  // row subtiles -> 8*4 = 32 v8f = 256 VGPRs.
  v8f acc[8][SUB];
#pragma unroll
  for (int j = 0; j < 8; ++j)
#pragma unroll
    for (int s = 0; s < SUB; ++s) acc[j][s] = vzero;

  // ---- Phase 0: stage x tile (f32 global -> bf16 A-fragment layout in LDS)
  // A-fragment mapping (16x32 bf16): lane' = row + 16*g, g = (K>>4)&1,
  // half index h = K&15. One float4 -> one 8 B LDS write.
  for (int f = tid; f < MT * (IN_CH / 4); f += 256) {
    int m = f >> 8;                 // IN_CH/4 == 256 float4 per row
    int k = (f & 255) << 2;
    v4f xv = *(const v4f*)(x + (size_t)(mbase + m) * IN_CH + k);
    int s   = m >> 4;
    int row = m & 15;
    int kb  = k >> 5;
    int g   = (k >> 4) & 1;
    int h   = k & 15;
    int lp  = row + 16 * g;
    v4bf o = { tobf(xv.x), tobf(xv.y), tobf(xv.z), tobf(xv.w) };
    *(v4bf*)(smem + ((kb * SUB + s) * 32 + lp) * 32 + h * 2) = o;
  }
  __syncthreads();

  // ---- Hidden-blocked main loop
  for (int hb = 0; hb < NHB; ++hb) {
    // Phase A: h_blk[64, HB] = relu(x . W1[hb]^T + b1) -> bf16 A-frags in LDS
    for (int i = 0; i < (HB / 16) / NWAVE; ++i) {   // 4 N-tiles per wave
      const int nt = wv + NWAVE * i;                // 0..31 within block
      v8f c[SUB];
#pragma unroll
      for (int s = 0; s < SUB; ++s) c[s] = vzero;
      const __bf16* wrow =
          w1b + (size_t)(hb * HB + nt * 16 + bcol) * IN_CH + bg * 16;
#pragma unroll 2
      for (int kb = 0; kb < IN_CH / 32; ++kb) {
        v16bf bf = *(const v16bf*)(wrow + kb * 32);
        __builtin_prefetch(wrow + kb * 32 + 128, 0, 3);  // global_prefetch_b8
        const char* ab = smem + kb * (SUB * FRAG_B) + lane * 32;
        v16bf a[SUB];
#pragma unroll
        for (int s = 0; s < SUB; ++s)
          a[s] = *(const v16bf*)(ab + s * FRAG_B);
#pragma unroll
        for (int s = 0; s < SUB; ++s)
          c[s] = wmma_bf16(a[s], bf, c[s]);
      }
      // bias + relu + store as layer-2 A-fragments.
      // C layout: elem r => M = r + 8*bg, N = bcol. Block-hidden index
      // K2 = nt*16+bcol => kb2 = nt>>1, lane' = (r+8*bg) + 16*(nt&1),
      // h' = bcol.
      float bias = b1[hb * HB + nt * 16 + bcol];
      int kb2 = nt >> 1;
      int hp  = bcol;
      int gpr = 16 * (nt & 1);
#pragma unroll
      for (int s = 0; s < SUB; ++s) {
#pragma unroll
        for (int r = 0; r < 8; ++r) {
          int lp = (r + 8 * bg) + gpr;
          *(__bf16*)(smem + HBK_OFF + ((kb2 * SUB + s) * 32 + lp) * 32 +
                     hp * 2) = tobf(fmaxf(c[s][r] + bias, 0.0f));
        }
      }
    }
    __syncthreads();

    // Phase B: acc += h_blk . W2[:, hb]^T   (A from LDS, B from W2 bf16)
#pragma unroll
    for (int j = 0; j < 8; ++j) {
      const int ot = wv + NWAVE * j;
      const __bf16* wrow =
          w2b + (size_t)(ot * 16 + bcol) * HID_CH + hb * HB + bg * 16;
#pragma unroll 2
      for (int kb = 0; kb < HB / 32; ++kb) {
        v16bf bf = *(const v16bf*)(wrow + kb * 32);
        __builtin_prefetch(wrow + kb * 32 + 128, 0, 3);
        const char* ab = smem + HBK_OFF + kb * (SUB * FRAG_B) + lane * 32;
        v16bf a[SUB];
#pragma unroll
        for (int s = 0; s < SUB; ++s)
          a[s] = *(const v16bf*)(ab + s * FRAG_B);
#pragma unroll
        for (int s = 0; s < SUB; ++s)
          acc[j][s] = wmma_bf16(a[s], bf, acc[j][s]);
      }
    }
    __syncthreads();
  }

  // ---- Epilogue: bias + store f32 output
#pragma unroll
  for (int j = 0; j < 8; ++j) {
    const int ot = wv + NWAVE * j;
    float bias = b2[ot * 16 + bcol];
#pragma unroll
    for (int s = 0; s < SUB; ++s) {
#pragma unroll
      for (int r = 0; r < 8; ++r) {
        int row = s * 16 + r + 8 * bg;
        out[(size_t)(mbase + row) * OUT_CH + ot * 16 + bcol] =
            acc[j][s][r] + bias;
      }
    }
  }
}

// ---------------------------------------------------------------------------
extern "C" void kernel_launch(void* const* d_in, const int* in_sizes, int n_in,
                              void* d_out, int out_size, void* d_ws, size_t ws_size,
                              hipStream_t stream) {
  (void)n_in; (void)out_size; (void)ws_size;
  const float* x  = (const float*)d_in[0];
  const float* W1 = (const float*)d_in[1];
  const float* b1 = (const float*)d_in[2];
  const float* W2 = (const float*)d_in[3];
  const float* b2 = (const float*)d_in[4];
  float* out = (float*)d_out;

  __bf16* w1b = (__bf16*)d_ws;                          // 8 MB
  __bf16* w2b = w1b + (size_t)HID_CH * IN_CH;           // 8 MB

  const int M = in_sizes[0] / IN_CH;                    // 8*4096 = 32768

  // Convert both weight matrices (4M elements each) to bf16.
  mlp_cvt_weights<<<(HID_CH * IN_CH) / 4 / 256, 256, 0, stream>>>(W1, W2, w1b, w2b);

  // Fused MLP: 512 workgroups x 256 threads, 192 KB dynamic LDS each.
  mlp_fused<<<M / MT, 256, LDS_BYTES, stream>>>(x, w1b, b1, w2b, b2, out);
}
